// BrainGFM_44178033607223
// MI455X (gfx1250) — matrix operand order repaired and checked by reading, not compiled, not run on hardware
//
#include <hip/hip_runtime.h>
#include <hip/hip_bf16.h>
#include <cmath>

typedef __attribute__((ext_vector_type(16))) _Float16 v16h;
typedef __attribute__((ext_vector_type(2)))  _Float16 h2f;
typedef __attribute__((ext_vector_type(8)))  float    v8f;

#define BM 64
#define BN 64
#define BK 32
#define LDS_STRIDE 40   // halves; 80B row stride -> 16B aligned frags, conflict-free (bank step 20)
#define ASIZE (BM * LDS_STRIDE)
#define BSIZE (BN * LDS_STRIDE)

// ---- panel staging: fast path (uniform, branch-free, float2 loads) or clamped edge path ----
__device__ __forceinline__ void stage_panels(
    const float* __restrict__ Ap, int lda,
    const float* __restrict__ Bp, int ldb, int bT,
    int bm0, int bn0, int M, int N, int K, int k0, int tid,
    _Float16* __restrict__ dA, _Float16* __restrict__ dB)
{
    const bool fullK = (k0 + BK <= K);
    // ---------------- A panel: 64 rows x 32 k -> dA[m][k] ----------------
    if ((bm0 + BM <= M) && fullK) {
        #pragma unroll
        for (int it = 0; it < 4; ++it) {
            int r  = (tid >> 4) + it * 16;       // 0..63
            int k2 = (tid & 15) * 2;             // 0,2,..,30
            float2 v = *reinterpret_cast<const float2*>(
                &Ap[(long long)(bm0 + r) * lda + k0 + k2]);
            h2f pv; pv[0] = (_Float16)v.x; pv[1] = (_Float16)v.y;
            *reinterpret_cast<h2f*>(&dA[r * LDS_STRIDE + k2]) = pv;
        }
    } else {
        #pragma unroll
        for (int it = 0; it < 4; ++it) {
            int r  = (tid >> 4) + it * 16;
            int k2 = (tid & 15) * 2;
            int gm = bm0 + r, gk = k0 + k2;
            int gmc = gm < M ? gm : M - 1;
            int gk0 = gk < K ? gk : K - 1;
            int gk1 = gk + 1 < K ? gk + 1 : K - 1;
            float v0 = Ap[(long long)gmc * lda + gk0];
            float v1 = Ap[(long long)gmc * lda + gk1];
            if (gm >= M || gk     >= K) v0 = 0.f;
            if (gm >= M || gk + 1 >= K) v1 = 0.f;
            h2f pv; pv[0] = (_Float16)v0; pv[1] = (_Float16)v1;
            *reinterpret_cast<h2f*>(&dA[r * LDS_STRIDE + k2]) = pv;
        }
    }
    // ---------------- B panel -> dB[n][k] (staged K-contiguous per column) ----------------
    if (bT) {   // B stored [N,K]: same access shape as A
        if ((bn0 + BN <= N) && fullK) {
            #pragma unroll
            for (int it = 0; it < 4; ++it) {
                int n  = (tid >> 4) + it * 16;
                int k2 = (tid & 15) * 2;
                float2 v = *reinterpret_cast<const float2*>(
                    &Bp[(long long)(bn0 + n) * ldb + k0 + k2]);
                h2f pv; pv[0] = (_Float16)v.x; pv[1] = (_Float16)v.y;
                *reinterpret_cast<h2f*>(&dB[n * LDS_STRIDE + k2]) = pv;
            }
        } else {
            #pragma unroll
            for (int it = 0; it < 4; ++it) {
                int n  = (tid >> 4) + it * 16;
                int k2 = (tid & 15) * 2;
                int gn = bn0 + n, gk = k0 + k2;
                int gnc = gn < N ? gn : N - 1;
                int gk0 = gk < K ? gk : K - 1;
                int gk1 = gk + 1 < K ? gk + 1 : K - 1;
                float v0 = Bp[(long long)gnc * ldb + gk0];
                float v1 = Bp[(long long)gnc * ldb + gk1];
                if (gn >= N || gk     >= K) v0 = 0.f;
                if (gn >= N || gk + 1 >= K) v1 = 0.f;
                h2f pv; pv[0] = (_Float16)v0; pv[1] = (_Float16)v1;
                *reinterpret_cast<h2f*>(&dB[n * LDS_STRIDE + k2]) = pv;
            }
        }
    } else {    // B stored [K,N]: read two k-rows (each coalesced over n), pack along k
        if ((bn0 + BN <= N) && fullK) {
            #pragma unroll
            for (int it = 0; it < 4; ++it) {
                int n  = tid & 63;
                int k2 = (tid >> 6) * 2 + it * 8;
                float v0 = Bp[(long long)(k0 + k2)     * ldb + bn0 + n];
                float v1 = Bp[(long long)(k0 + k2 + 1) * ldb + bn0 + n];
                h2f pv; pv[0] = (_Float16)v0; pv[1] = (_Float16)v1;
                *reinterpret_cast<h2f*>(&dB[n * LDS_STRIDE + k2]) = pv;
            }
        } else {
            #pragma unroll
            for (int it = 0; it < 4; ++it) {
                int n  = tid & 63;
                int k2 = (tid >> 6) * 2 + it * 8;
                int gn = bn0 + n, gk = k0 + k2;
                int gnc = gn < N ? gn : N - 1;
                int gk0 = gk < K ? gk : K - 1;
                int gk1 = gk + 1 < K ? gk + 1 : K - 1;
                float v0 = Bp[(long long)gk0 * ldb + gnc];
                float v1 = Bp[(long long)gk1 * ldb + gnc];
                if (gn >= N || gk     >= K) v0 = 0.f;
                if (gn >= N || gk + 1 >= K) v1 = 0.f;
                h2f pv; pv[0] = (_Float16)v0; pv[1] = (_Float16)v1;
                *reinterpret_cast<h2f*>(&dB[n * LDS_STRIDE + k2]) = pv;
            }
        }
    }
}

// ---------------------------------------------------------------------------
// Block-tiled batched WMMA GEMM with LDS double-buffering:
// C[bb] = act( alpha * A[bb] x B[bb] * scale + bias ),  bb = outer*nb2 + inner.
// 256 threads / 8 waves; block computes 64x64 of C; each wave a 32x16 strip
// = two v_wmma_f32_16x16x32_f16 per K-step sharing one B fragment.
// Next panel's global loads are issued before the current step's MMA work
// (ping-pong LDS buffers, one barrier per K-step) to hide memory latency.
// ---------------------------------------------------------------------------
__global__ __launch_bounds__(256) void wmma_gemm_kernel(
    const float* __restrict__ A, long long sA1, long long sA2, int nb2, int lda,
    const float* __restrict__ Bm, long long sB1, long long sB2, int ldb, int bT,
    const int* __restrict__ idxB,
    float* __restrict__ C, long long sC1, long long sC2, int ldc,
    int M, int N, int K,
    const float* __restrict__ bias, const float* __restrict__ scale,
    long long sSB, const int* __restrict__ idxSB,
    int relu, float alpha, int tilesM, int tilesN)
{
    __shared__ _Float16 smA[2 * ASIZE];
    __shared__ _Float16 smB[2 * BSIZE];

    int tid  = threadIdx.x;
    int wave = tid >> 5;
    int lane = tid & 31;

    int tilesPerBatch = tilesM * tilesN;
    int bid = blockIdx.x;
    int bb  = bid / tilesPerBatch;
    int t   = bid % tilesPerBatch;
    int tmi = t / tilesN, tni = t % tilesN;
    int outer = bb / nb2, inner = bb % nb2;

    const float* Ap = A + (long long)outer * sA1 + (long long)inner * sA2;
    long long bsel  = idxB ? (long long)idxB[outer] : (long long)outer;
    const float* Bp = Bm + bsel * sB1 + (long long)inner * sB2;
    float*       Cp = C + (long long)outer * sC1 + (long long)inner * sC2;
    long long sbsel = idxSB ? (long long)idxSB[outer] : 0;

    int bm0 = tmi * BM;
    int bn0 = tni * BN;

    int wm = wave >> 2;     // 32-row half
    int wn = wave & 3;      // 16-col strip

    // fragment addressing (ISA 16-bit A 16x32 / B 32x16 lane layouts)
    int frow = lane & 15;
    int hi   = lane >> 4;
    int kbA  = hi * 8;
    int kbB  = hi * 8;

    v8f acc0 = {0.f,0.f,0.f,0.f,0.f,0.f,0.f,0.f};
    v8f acc1 = {0.f,0.f,0.f,0.f,0.f,0.f,0.f,0.f};

    // prologue: stage first panel into buffer 0
    stage_panels(Ap, lda, Bp, ldb, bT, bm0, bn0, M, N, K, 0, tid, smA, smB);
    __syncthreads();

    int p = 0;
    for (int k0 = 0; k0 < K; k0 += BK) {
        const _Float16* curA = smA + p * ASIZE;
        const _Float16* curB = smB + p * BSIZE;

        // issue next panel's loads before this step's MMA (latency overlap)
        int kn = k0 + BK;
        if (kn < K)
            stage_panels(Ap, lda, Bp, ldb, bT, bm0, bn0, M, N, K, kn, tid,
                         smA + (1 - p) * ASIZE, smB + (1 - p) * BSIZE);

        // fragments from LDS (unconditional, 16B-aligned ds_load_b128)
        v16h a0, a1, bf;
        const _Float16* pa0 = &curA[(wm * 32 + frow) * LDS_STRIDE + kbA];
        const _Float16* pa1 = pa0 + 16 * LDS_STRIDE;
        const _Float16* pb  = &curB[(wn * 16 + frow) * LDS_STRIDE + kbB];
        #pragma unroll
        for (int i = 0; i < 8; ++i) {
            a0[i]   = pa0[i];
            a0[i+8] = pa0[16 + i];
            a1[i]   = pa1[i];
            a1[i+8] = pa1[16 + i];
        }
        #pragma unroll
        for (int i = 0; i < 8; ++i) { bf[i] = pb[i]; bf[i+8] = pb[16 + i]; }

        acc0 = __builtin_amdgcn_wmma_f32_16x16x32_f16(false, a0, false, bf, (short)0, acc0, false, false);
        acc1 = __builtin_amdgcn_wmma_f32_16x16x32_f16(false, a1, false, bf, (short)0, acc1, false, false);

        __syncthreads();
        p ^= 1;
    }

    // ---- epilogue ----
    const float* biasp  = bias  ? bias  + sbsel * sSB : nullptr;
    const float* scalep = scale ? scale + sbsel * sSB : nullptr;
    int n = bn0 + wn * 16 + (lane & 15);
    #pragma unroll
    for (int sub = 0; sub < 2; ++sub) {
        v8f& acc = sub ? acc1 : acc0;
        #pragma unroll
        for (int r = 0; r < 8; ++r) {
            int m = bm0 + wm * 32 + sub * 16 + r + hi * 8;
            if (m < M && n < N) {
                float v = acc[r] * alpha;
                if (scalep) v *= scalep[n];
                if (biasp)  v += biasp[n];
                if (relu)   v = fmaxf(v, 0.f);
                Cp[(long long)m * ldc + n] = v;
            }
        }
    }
}

// ---- adj row-normalize (adjn = adj / (rowsum+1e-6)), also writes rwse k=0 ----
__global__ void norm_adj_kernel(const float* __restrict__ adj, float* __restrict__ adjn,
                                float* __restrict__ rwse, int Ncol)
{
    long long row = blockIdx.x;               // b*200 + n
    const float* ap = adj + row * Ncol;
    int tid = threadIdx.x;
    __shared__ float red[256];
    float s = 0.f;
    for (int c = tid; c < Ncol; c += 256) s += ap[c];
    red[tid] = s; __syncthreads();
    for (int st = 128; st > 0; st >>= 1) { if (tid < st) red[tid] += red[tid + st]; __syncthreads(); }
    float inv = 1.f / (red[0] + 1e-6f);
    for (int c = tid; c < Ncol; c += 256) adjn[row * Ncol + c] = ap[c] * inv;
    if (tid == 0) {
        int n = (int)(row % Ncol);
        rwse[row * 5 + 0] = ap[n] * inv;
    }
}

__global__ void extract_diag_kernel(const float* __restrict__ rw, float* __restrict__ rwse,
                                    int k, int total, int Ncol)
{
    int id = blockIdx.x * blockDim.x + threadIdx.x;   // b*200 + n
    if (id < total) {
        int n = id % Ncol;
        rwse[(long long)id * 5 + k] = rw[(long long)id * Ncol + n];
    }
}

// ---- featp[b,n,c] = (c<200 ? nf : rwse) * prompt[n,c], c<205, ld=224 ----
__global__ void build_featp_kernel(const float* __restrict__ nf, const float* __restrict__ rwse,
                                   const float* __restrict__ prompt, float* __restrict__ featp,
                                   long long total)
{
    long long id = (long long)blockIdx.x * blockDim.x + threadIdx.x;
    if (id >= total) return;
    int c  = (int)(id % 205);
    long long rn = id / 205;                   // b*200 + n
    int n  = (int)(rn % 200);
    float v = (c < 200) ? nf[rn * 200 + c] : rwse[rn * 5 + (c - 200)];
    featp[rn * 224 + c] = v * prompt[n * 256 + c];
}

// ---- dis/parc token projection, broadcast into rows 0,1 of h ----
__global__ void dis_parc_kernel(const float* __restrict__ disE, const float* __restrict__ disW,
                                const float* __restrict__ disB, const float* __restrict__ parc,
                                const float* __restrict__ projW, const float* __restrict__ projB,
                                float* __restrict__ h, int B, int S, int H)
{
    int t = threadIdx.x;                        // 128 threads == H
    float d = disB[t];
    for (int k = 0; k < 768; ++k) d += disE[k] * disW[k * H + t];
    float p = projB[t];
    for (int k = 0; k < 256; ++k) p += parc[k] * projW[k * H + t];
    for (int b = 0; b < B; ++b) {
        h[((long long)b * S + 0) * H + t] = d;
        h[((long long)b * S + 1) * H + t] = p;
    }
}

// ---- row softmax over ncols (fp32) ----
__global__ void softmax_kernel(float* __restrict__ Sc, int ncols, int ld)
{
    long long row = blockIdx.x;
    float* p = Sc + row * ld;
    int tid = threadIdx.x;
    __shared__ float red[256];
    float m = -1e30f;
    for (int c = tid; c < ncols; c += 256) m = fmaxf(m, p[c]);
    red[tid] = m; __syncthreads();
    for (int st = 128; st > 0; st >>= 1) { if (tid < st) red[tid] = fmaxf(red[tid], red[tid+st]); __syncthreads(); }
    m = red[0]; __syncthreads();
    float s = 0.f;
    for (int c = tid; c < ncols; c += 256) { float e = __expf(p[c] - m); p[c] = e; s += e; }
    red[tid] = s; __syncthreads();
    for (int st = 128; st > 0; st >>= 1) { if (tid < st) red[tid] += red[tid+st]; __syncthreads(); }
    float inv = 1.f / red[0];
    for (int c = tid; c < ncols; c += 256) p[c] *= inv;
}

// ---- h = LayerNorm(h + a) * g + b, H=128, one row per block ----
__global__ void ln_residual_kernel(float* __restrict__ h, const float* __restrict__ a,
                                   const float* __restrict__ g, const float* __restrict__ b, int H)
{
    long long row = blockIdx.x;
    int t = threadIdx.x;
    float* hp = h + row * H;
    float v = hp[t] + a[row * H + t];
    __shared__ float red[128];
    red[t] = v; __syncthreads();
    for (int st = 64; st > 0; st >>= 1) { if (t < st) red[t] += red[t+st]; __syncthreads(); }
    float mean = red[0] / H; __syncthreads();
    float d = v - mean;
    red[t] = d * d; __syncthreads();
    for (int st = 64; st > 0; st >>= 1) { if (t < st) red[t] += red[t+st]; __syncthreads(); }
    float var = red[0] / H;
    hp[t] = d * rsqrtf(var + 1e-5f) * g[t] + b[t];
}

// ---- top-1 router: mean over rows, logits = m @ rW + rb, argmax (first max) ----
__global__ void route_top1_kernel(const float* __restrict__ x, long long strideB, int rows, int H,
                                  const float* __restrict__ rW, const float* __restrict__ rb,
                                  int E, int* __restrict__ top1)
{
    int b = blockIdx.x;
    int t = threadIdx.x;                        // 128
    const float* xp = x + (long long)b * strideB;
    float s = 0.f;
    for (int r = 0; r < rows; ++r) s += xp[(long long)r * H + t];
    __shared__ float m[128];
    __shared__ float logit[8];
    m[t] = s / rows; __syncthreads();
    if (t < E) {
        float acc = rb[t];
        for (int hh = 0; hh < H; ++hh) acc += m[hh] * rW[hh * E + t];
        logit[t] = acc;
    }
    __syncthreads();
    if (t == 0) {
        int best = 0; float bv = logit[0];
        for (int e = 1; e < E; ++e) if (logit[e] > bv) { bv = logit[e]; best = e; }
        top1[b] = best;
    }
}

// ---- g = mean over rows of gout ----
__global__ void mean_pool_kernel(const float* __restrict__ gin, float* __restrict__ out,
                                 int rows, int H)
{
    int b = blockIdx.x, t = threadIdx.x;
    float s = 0.f;
    for (int r = 0; r < rows; ++r) s += gin[((long long)b * rows + r) * H + t];
    out[(long long)b * H + t] = s / rows;
}

// ---------------------------------------------------------------------------
static inline void gemm(hipStream_t st,
    const float* A, long long sA1, long long sA2, int nb2, int lda,
    const float* B, long long sB1, long long sB2, int ldb, int bT, const int* idxB,
    float* C, long long sC1, long long sC2, int ldc,
    int M, int N, int K, int batches,
    const float* bias = nullptr, const float* scale = nullptr, long long sSB = 0,
    const int* idxSB = nullptr, int relu = 0, float alpha = 1.f)
{
    int tm = (M + BM - 1) / BM, tn = (N + BN - 1) / BN;
    int blocks = batches * tm * tn;
    wmma_gemm_kernel<<<blocks, 256, 0, st>>>(A, sA1, sA2, nb2, lda,
        B, sB1, sB2, ldb, bT, idxB, C, sC1, sC2, ldc, M, N, K,
        bias, scale, sSB, idxSB, relu, alpha, tm, tn);
}

extern "C" void kernel_launch(void* const* d_in, const int* in_sizes, int n_in,
                              void* d_out, int out_size, void* d_ws, size_t ws_size,
                              hipStream_t stream)
{
    (void)in_sizes; (void)n_in; (void)out_size; (void)ws_size;
    const int B = 128, N = 200, H = 128, E = 4, FF = 512, S = 202, NH = 8, DH = 16;
    typedef long long ll;

    const float* nf    = (const float*)d_in[0];
    const float* adj   = (const float*)d_in[1];
    const float* prompt= (const float*)d_in[2];
    const float* parc  = (const float*)d_in[3];
    const float* disE  = (const float*)d_in[4];
    const float* projW = (const float*)d_in[5];
    const float* projB = (const float*)d_in[6];
    const float* disW  = (const float*)d_in[7];
    const float* disB  = (const float*)d_in[8];
    const float* Wqkv  = (const float*)d_in[9];
    const float* bqkv  = (const float*)d_in[10];
    const float* Wo    = (const float*)d_in[11];
    const float* bo    = (const float*)d_in[12];
    const float* ln1g  = (const float*)d_in[13];
    const float* ln1b  = (const float*)d_in[14];
    const float* ln2g  = (const float*)d_in[15];
    const float* ln2b  = (const float*)d_in[16];
    const float* frW   = (const float*)d_in[17];
    const float* frb   = (const float*)d_in[18];
    const float* fW1   = (const float*)d_in[19];
    const float* fb1   = (const float*)d_in[20];
    const float* fW2   = (const float*)d_in[21];
    const float* fb2   = (const float*)d_in[22];
    const float* grW   = (const float*)d_in[23];
    const float* grb   = (const float*)d_in[24];
    const float* gWt   = (const float*)d_in[25];
    const float* bng   = (const float*)d_in[26];
    const float* bnb   = (const float*)d_in[27];
    float* out = (float*)d_out;
    float* wsf = (float*)d_ws;

    // workspace layout (floats), with dead-region reuse
    size_t o = 0;
    auto take = [&](size_t n) { size_t r = o; o += (n + 63) & ~(size_t)63; return r; };
    size_t o_adjn  = take((size_t)B * N * N);     // dead after RWSE -> reused by ffnh
    size_t o_rwA   = take((size_t)B * N * N);
    size_t o_rwB   = take((size_t)B * N * N);
    size_t o_rwse  = take((size_t)B * N * 5);
    size_t o_featp = take((size_t)B * N * 224);   // dead after proj -> reused by scores
    size_t o_h     = take((size_t)B * S * H);
    size_t o_qkv   = take((size_t)B * S * 3 * H);
    size_t o_attno = take((size_t)B * S * H);
    size_t o_awo   = take((size_t)B * S * H);
    size_t o_sup   = take((size_t)B * N * H);
    size_t o_gout  = take((size_t)B * N * H);
    size_t o_topF  = take(128);
    size_t o_topG  = take(128);

    float* adjn  = wsf + o_adjn;
    float* rwA   = wsf + o_rwA;
    float* rwB   = wsf + o_rwB;
    float* rwse  = wsf + o_rwse;
    float* featp = wsf + o_featp;
    float* h     = wsf + o_h;
    float* qkvb  = wsf + o_qkv;
    float* attno = wsf + o_attno;
    float* awo   = wsf + o_awo;
    float* sup   = wsf + o_sup;
    float* gout  = wsf + o_gout;
    float* ffnh   = wsf + o_adjn;   // [B,S,FF] = 13.24M floats <= 15.36M (adjn+rwA+rwB)
    float* scores = wsf + o_featp;  // chunk of 16 batches x 8 heads x 202 x 202
    int* topF = (int*)(wsf + o_topF);
    int* topG = (int*)(wsf + o_topG);

    // ---------------- Stage 1: RWSE ----------------
    norm_adj_kernel<<<B * N, 256, 0, stream>>>(adj, adjn, rwse, N);
    gemm(stream, adjn, (ll)N*N, 0, 1, N, adjn, (ll)N*N, 0, N, 0, nullptr,
         rwA, (ll)N*N, 0, N, N, N, N, B);
    extract_diag_kernel<<<(B*N + 255)/256, 256, 0, stream>>>(rwA, rwse, 1, B*N, N);
    gemm(stream, rwA, (ll)N*N, 0, 1, N, adjn, (ll)N*N, 0, N, 0, nullptr,
         rwB, (ll)N*N, 0, N, N, N, N, B);
    extract_diag_kernel<<<(B*N + 255)/256, 256, 0, stream>>>(rwB, rwse, 2, B*N, N);
    gemm(stream, rwB, (ll)N*N, 0, 1, N, adjn, (ll)N*N, 0, N, 0, nullptr,
         rwA, (ll)N*N, 0, N, N, N, N, B);
    extract_diag_kernel<<<(B*N + 255)/256, 256, 0, stream>>>(rwA, rwse, 3, B*N, N);
    gemm(stream, rwA, (ll)N*N, 0, 1, N, adjn, (ll)N*N, 0, N, 0, nullptr,
         rwB, (ll)N*N, 0, N, N, N, N, B);
    extract_diag_kernel<<<(B*N + 255)/256, 256, 0, stream>>>(rwB, rwse, 4, B*N, N);

    // ---------------- Stage 2: feature build + projection ----------------
    {
        long long total = (ll)B * N * 205;
        build_featp_kernel<<<(int)((total + 255)/256), 256, 0, stream>>>(nf, rwse, prompt, featp, total);
    }
    gemm(stream, featp, (ll)N*224, 0, 1, 224, projW, 0, 0, H, 0, nullptr,
         h + 2*H, (ll)S*H, 0, H, N, H, 205, B, projB);
    dis_parc_kernel<<<1, 128, 0, stream>>>(disE, disW, disB, parc, projW, projB, h, B, S, H);

    // ---------------- Stage 3: transformer blocks (outer i=1 only; i=0 is dead) ----------------
    const int CB = 16, NCH = B / CB;   // attention batch chunks
    for (int j = 0; j < 2; ++j) {
        const int ij = 1 * 2 + j;      // weight index [i=1][j]
        // QKV: [B,S,H] @ Wqkv^T -> [B,S,3H]
        gemm(stream, h, (ll)S*H, 0, 1, H, Wqkv + (ll)ij*3*H*H, 0, 0, H, 1, nullptr,
             qkvb, (ll)S*3*H, 0, 3*H, S, 3*H, H, B, bqkv + (ll)ij*3*H);
        // attention, chunked over batch
        for (int c = 0; c < NCH; ++c) {
            int b0 = c * CB;
            const float* qb = qkvb + (ll)b0 * S * 3 * H;
            // scores = (q @ k^T) * dh^-0.5   [CB*NH, S, S]
            gemm(stream, qb, (ll)S*3*H, 16, NH, 3*H,
                 qb + H, (ll)S*3*H, 16, 3*H, 1, nullptr,
                 scores, (ll)NH*S*S, (ll)S*S, S,
                 S, S, DH, CB*NH, nullptr, nullptr, 0, nullptr, 0, 0.25f);
            softmax_kernel<<<CB*NH*S, 256, 0, stream>>>(scores, S, S);
            // o = P @ v -> attno[b, s, head*16 + d]
            gemm(stream, scores, (ll)NH*S*S, (ll)S*S, NH, S,
                 qb + 2*H, (ll)S*3*H, 16, 3*H, 0, nullptr,
                 attno + (ll)b0*S*H, (ll)S*H, 16, H,
                 S, DH, S, CB*NH);
        }
        // Wo projection
        gemm(stream, attno, (ll)S*H, 0, 1, H, Wo + (ll)ij*H*H, 0, 0, H, 1, nullptr,
             awo, (ll)S*H, 0, H, S, H, H, B, bo + (ll)ij*H);
        // h = LN(h + a)
        ln_residual_kernel<<<B*S, 128, 0, stream>>>(h, awo, ln1g + (ll)ij*H, ln1b + (ll)ij*H, H);
        // MoE FFN: route on post-LN h (all 202 tokens)
        route_top1_kernel<<<B, 128, 0, stream>>>(h, (ll)S*H, S, H,
                                                 frW + (ll)ij*H*E, frb + (ll)ij*E, E, topF);
        gemm(stream, h, (ll)S*H, 0, 1, H,
             fW1 + (ll)ij*E*H*FF, (ll)H*FF, 0, FF, 0, topF,
             ffnh, (ll)S*FF, 0, FF, S, FF, H, B,
             fb1 + (ll)ij*E*FF, nullptr, FF, topF, 1);
        gemm(stream, ffnh, (ll)S*FF, 0, 1, FF,
             fW2 + (ll)ij*E*FF*H, (ll)FF*H, 0, H, 0, topF,
             awo, (ll)S*H, 0, H, S, H, FF, B,
             fb2 + (ll)ij*E*H, nullptr, H, topF, 0);
        ln_residual_kernel<<<B*S, 128, 0, stream>>>(h, awo, ln2g + (ll)ij*H, ln2b + (ll)ij*H, H);
    }

    // ---------------- Stage 4: MoE GCN (i=1) + readout ----------------
    route_top1_kernel<<<B, 128, 0, stream>>>(h + 2*H, (ll)S*H, N, H,
                                             grW + (ll)1*H*E, grb + (ll)1*E, E, topG);
    // support = x_nodes @ W[e]
    gemm(stream, h + 2*H, (ll)S*H, 0, 1, H,
         gWt + (ll)1*E*H*H, (ll)H*H, 0, H, 0, topG,
         sup, (ll)N*H, 0, H, N, H, H, B);
    // out = relu( (adj @ support) * rsqrt(1+eps) * gam[e] + bet[e] )
    gemm(stream, adj, (ll)N*N, 0, 1, N,
         sup, (ll)N*H, 0, H, 0, nullptr,
         gout, (ll)N*H, 0, H, N, H, N, B,
         bnb + (ll)1*E*H, bng + (ll)1*E*H, H, topG, 1,
         1.0f / sqrtf(1.0f + 1e-5f));
    mean_pool_kernel<<<B, 128, 0, stream>>>(gout, out, N, H);
}
